// CfC_net_84963043049828
// MI455X (gfx1250) — matrix operand order repaired and checked by reading, not compiled, not run on hardware
//
#include <hip/hip_runtime.h>
#include <hip/hip_bf16.h>
#include <math.h>

typedef _Float16 half_t;
typedef __attribute__((ext_vector_type(16))) _Float16 v16h;
typedef __attribute__((ext_vector_type(8)))  _Float16 v8h;
typedef __attribute__((ext_vector_type(8)))  float    v8f;

#define UNITS   64
#define IN_DIM  64
#define BB      128
#define NACT    18
#define T_STEPS 1024
#define MTILE   16
#define KPAD    136   // row stride (halves) for K=128 tiles, 272B = 16B aligned
#define OKPAD   72    // row stride (halves) for K=64 tiles, 144B = 16B aligned

// ---- branch-free transcendentals: one v_exp_f32 + one v_rcp_f32 each ----
__device__ __forceinline__ float fast_rcp(float x) {
    return __builtin_amdgcn_rcpf(x);
}
// 1.7159*tanh(0.666*x) = 1.7159 - 3.4318 / (exp(1.332*x) + 1)
__device__ __forceinline__ float lecun_tanh(float x) {
    float e = __expf(1.332f * x);
    return 1.7159f - 3.4318f * fast_rcp(e + 1.0f);
}
// sigmoid(b - a) = 1 / (1 + exp(a - b))
__device__ __forceinline__ float sigmoid_diff(float a, float b) {
    return fast_rcp(1.0f + __expf(a - b));
}

// A-matrix fragment (16x32 f16) per CDNA5 ISA layout:
// lane<16: halves 0..7 -> K=kt*32+0..7,   halves 8..15 -> K=kt*32+16..23
// lane>=16: halves 0..7 -> K=kt*32+8..15, halves 8..15 -> K=kt*32+24..31 ; M=lane&15
__device__ __forceinline__ v16h load_Afrag(const half_t* base, int stride, int kt, int lane) {
    int m = lane & 15;
    int kbase = kt * 32 + ((lane >> 4) << 3);
    const half_t* p = base + m * stride + kbase;
    v8h lo = *(const v8h*)(p);
    v8h hi = *(const v8h*)(p + 16);
    v16h a;
#pragma unroll
    for (int i = 0; i < 8; ++i) { a[i] = lo[i]; a[i + 8] = hi[i]; }
    return a;
}

// B-matrix fragment (32x16 f16): lane 0-15 hold K=kt*32+0..7,16..23 for N=lane,
// lane 16-31 hold K=kt*32+8..15,24..31 for N=lane-16. Weights stored transposed:
// wT[n][k] so each lane reads two runs of 8 contiguous halves.
__device__ __forceinline__ v16h load_Bfrag(const half_t* wT, int stride, int n0, int kt, int lane) {
    int n = n0 + (lane & 15);
    int kbase = kt * 32 + ((lane >> 4) << 3);
    const half_t* p = wT + n * stride + kbase;
    v8h lo = *(const v8h*)(p);
    v8h hi = *(const v8h*)(p + 16);
    v16h b;
#pragma unroll
    for (int i = 0; i < 8; ++i) { b[i] = lo[i]; b[i + 8] = hi[i]; }
    return b;
}

#define WMMA_F16(A, Bf, C) \
    __builtin_amdgcn_wmma_f32_16x16x32_f16(false, (A), false, (Bf), (short)0, (C), false, false)

__global__ __launch_bounds__(128) void cfc_scan_kernel(
    const float* __restrict__ x,     const float* __restrict__ h0,
    const float* __restrict__ bb_w,  const float* __restrict__ bb_b,
    const float* __restrict__ ff1_w, const float* __restrict__ ff1_b,
    const float* __restrict__ ff2_w, const float* __restrict__ ff2_b,
    const float* __restrict__ ta_w,  const float* __restrict__ ta_b,
    const float* __restrict__ tb_w,  const float* __restrict__ tb_b,
    const float* __restrict__ out_w, const float* __restrict__ out_b,
    float* __restrict__ y)
{
    // ---- LDS-resident state & weights (~117 KB of the WGP's 320 KB) ----
    __shared__ half_t zin[MTILE][KPAD];          // cols 0..63: x_t(f16), 64..127: h(f16)
    __shared__ half_t zbb[MTILE][KPAD];          // backbone activations
    __shared__ half_t bbT[BB][KPAD];             // bbT[n][k] = bb_w[k][n]
    __shared__ half_t headT[4][UNITS][KPAD];     // ff1, ff2, ta, tb (transposed)
    __shared__ half_t outT[32][OKPAD];           // outT[n][k] = out_w[k][n], n>=18 zero
    __shared__ float  bb_bias[BB];
    __shared__ float  head_bias[4][UNITS];
    __shared__ float  out_bias[32];

    const int tid  = threadIdx.x;
    const int lane = tid & 31;
    const int wv   = tid >> 5;                   // wave 0..3
    const int b0   = blockIdx.x * MTILE;         // batch tile origin

    // ---- one-time: convert weights f32 -> f16 transposed into LDS ----
    for (int i = tid; i < BB * BB; i += 128) {
        int k = i >> 7, n = i & 127;
        bbT[n][k] = (half_t)bb_w[i];
    }
    {
        const float* hw[4] = { ff1_w, ff2_w, ta_w, tb_w };
        const float* hb[4] = { ff1_b, ff2_b, ta_b, tb_b };
#pragma unroll
        for (int hd = 0; hd < 4; ++hd) {
            for (int i = tid; i < BB * UNITS; i += 128) {
                int k = i >> 6, n = i & 63;
                headT[hd][n][k] = (half_t)hw[hd][i];
            }
            if (tid < UNITS) head_bias[hd][tid] = hb[hd][tid];
        }
    }
    for (int i = tid; i < 32 * UNITS; i += 128) {
        int n = i >> 6, k = i & 63;
        outT[n][k] = (n < NACT) ? (half_t)out_w[k * NACT + n] : (half_t)0.f;
    }
    if (tid < BB) bb_bias[tid] = bb_b[tid];
    if (tid < 32) out_bias[tid] = (tid < NACT) ? out_b[tid] : 0.f;
    // h0 -> zin[:,64:128]
    for (int i = tid; i < MTILE * UNITS; i += 128) {
        int m = i >> 6, j = i & 63;
        zin[m][UNITS + j] = (half_t)h0[(size_t)(b0 + m) * UNITS + j];
    }
    __syncthreads();

    const int xm = tid >> 3;                     // x-tile row this thread loads
    const int xf = (tid & 7) * 8;                // x-tile col base

    for (int t = 0; t < T_STEPS; ++t) {
        // ---------- stage 0: x_t -> zin[:,0:64] ----------
        {
            const float* xp = x + ((size_t)(b0 + xm) * T_STEPS + t) * IN_DIM + xf;
            float4 v0 = *(const float4*)(xp);
            float4 v1 = *(const float4*)(xp + 4);
            v8h hx;
            hx[0] = (half_t)v0.x; hx[1] = (half_t)v0.y; hx[2] = (half_t)v0.z; hx[3] = (half_t)v0.w;
            hx[4] = (half_t)v1.x; hx[5] = (half_t)v1.y; hx[6] = (half_t)v1.z; hx[7] = (half_t)v1.w;
            *(v8h*)&zin[xm][xf] = hx;
            if (t + 1 < T_STEPS) __builtin_prefetch(xp + IN_DIM, 0, 0);  // next timestep
        }
        __syncthreads();   // makes x_t writes AND previous step's h writes visible

        // ---------- stage 1: backbone  z = lecun_tanh([x,h] @ bb_w + b) ----------
        v16h a0 = load_Afrag(&zin[0][0], KPAD, 0, lane);
        v16h a1 = load_Afrag(&zin[0][0], KPAD, 1, lane);
        v16h a2 = load_Afrag(&zin[0][0], KPAD, 2, lane);   // == h rows, K 0..31
        v16h a3 = load_Afrag(&zin[0][0], KPAD, 3, lane);   // == h rows, K 32..63

#pragma unroll
        for (int c = 0; c < 2; ++c) {
            int n0 = (2 * wv + c) * 16;
            float bias = bb_bias[n0 + (lane & 15)];
            v8f acc = { bias, bias, bias, bias, bias, bias, bias, bias };
            v16h b;
            b = load_Bfrag(&bbT[0][0], KPAD, n0, 0, lane); acc = WMMA_F16(a0, b, acc);
            b = load_Bfrag(&bbT[0][0], KPAD, n0, 1, lane); acc = WMMA_F16(a1, b, acc);
            b = load_Bfrag(&bbT[0][0], KPAD, n0, 2, lane); acc = WMMA_F16(a2, b, acc);
            b = load_Bfrag(&bbT[0][0], KPAD, n0, 3, lane); acc = WMMA_F16(a3, b, acc);
            int mrow = (lane >> 4) * 8;
            int n = n0 + (lane & 15);
#pragma unroll
            for (int r = 0; r < 8; ++r)
                zbb[mrow + r][n] = (half_t)lecun_tanh(acc[r]);
        }

        // deferred output projection: y[t-1] = h_t @ out_w + out_b (h_t == a2,a3)
        if (wv < 2 && t > 0) {
            int n0 = wv * 16;
            float ob = out_bias[n0 + (lane & 15)];
            v8f acc = { ob, ob, ob, ob, ob, ob, ob, ob };
            v16h b;
            b = load_Bfrag(&outT[0][0], OKPAD, n0, 0, lane); acc = WMMA_F16(a2, b, acc);
            b = load_Bfrag(&outT[0][0], OKPAD, n0, 1, lane); acc = WMMA_F16(a3, b, acc);
            int n = n0 + (lane & 15);
            if (n < NACT) {
                int mrow = (lane >> 4) * 8;
#pragma unroll
                for (int r = 0; r < 8; ++r)
                    y[((size_t)(b0 + mrow + r) * T_STEPS + (t - 1)) * NACT + n] = acc[r];
            }
        }
        __syncthreads();

        // ---------- stage 2: four heads + CfC gate fuse ----------
        v16h za0 = load_Afrag(&zbb[0][0], KPAD, 0, lane);
        v16h za1 = load_Afrag(&zbb[0][0], KPAD, 1, lane);
        v16h za2 = load_Afrag(&zbb[0][0], KPAD, 2, lane);
        v16h za3 = load_Afrag(&zbb[0][0], KPAD, 3, lane);
        const int n0h = wv * 16;
        const int nn  = n0h + (lane & 15);
        v8f hacc[4];
#pragma unroll
        for (int hd = 0; hd < 4; ++hd) {
            float bias = head_bias[hd][nn];
            v8f acc = { bias, bias, bias, bias, bias, bias, bias, bias };
            v16h b;
            b = load_Bfrag(&headT[hd][0][0], KPAD, n0h, 0, lane); acc = WMMA_F16(za0, b, acc);
            b = load_Bfrag(&headT[hd][0][0], KPAD, n0h, 1, lane); acc = WMMA_F16(za1, b, acc);
            b = load_Bfrag(&headT[hd][0][0], KPAD, n0h, 2, lane); acc = WMMA_F16(za2, b, acc);
            b = load_Bfrag(&headT[hd][0][0], KPAD, n0h, 3, lane); acc = WMMA_F16(za3, b, acc);
            hacc[hd] = acc;
        }
        {
            int mrow = (lane >> 4) * 8;
#pragma unroll
            for (int r = 0; r < 8; ++r) {
                float f1 = lecun_tanh(hacc[0][r]);
                float f2 = lecun_tanh(hacc[1][r]);
                float s  = sigmoid_diff(hacc[2][r], hacc[3][r]);  // sigmoid(tb - ta)
                float hn = f1 + s * (f2 - f1);
                zin[mrow + r][UNITS + nn] = (half_t)hn;   // h for next step
            }
        }
        // no barrier needed: next stage 0 writes disjoint zin columns (0..63)
    }

    // ---------- flush y[T-1] ----------
    __syncthreads();
    if (wv < 2) {
        v16h a2 = load_Afrag(&zin[0][0], KPAD, 2, lane);
        v16h a3 = load_Afrag(&zin[0][0], KPAD, 3, lane);
        int n0 = wv * 16;
        float ob = out_bias[n0 + (lane & 15)];
        v8f acc = { ob, ob, ob, ob, ob, ob, ob, ob };
        v16h b;
        b = load_Bfrag(&outT[0][0], OKPAD, n0, 0, lane); acc = WMMA_F16(a2, b, acc);
        b = load_Bfrag(&outT[0][0], OKPAD, n0, 1, lane); acc = WMMA_F16(a3, b, acc);
        int n = n0 + (lane & 15);
        if (n < NACT) {
            int mrow = (lane >> 4) * 8;
#pragma unroll
            for (int r = 0; r < 8; ++r)
                y[((size_t)(b0 + mrow + r) * T_STEPS + (T_STEPS - 1)) * NACT + n] = acc[r];
        }
    }
}

extern "C" void kernel_launch(void* const* d_in, const int* in_sizes, int n_in,
                              void* d_out, int out_size, void* d_ws, size_t ws_size,
                              hipStream_t stream) {
    (void)in_sizes; (void)n_in; (void)out_size; (void)d_ws; (void)ws_size;
    const float* x     = (const float*)d_in[0];
    const float* h0    = (const float*)d_in[1];
    const float* bb_w  = (const float*)d_in[2];
    const float* bb_b  = (const float*)d_in[3];
    const float* ff1_w = (const float*)d_in[4];
    const float* ff1_b = (const float*)d_in[5];
    const float* ff2_w = (const float*)d_in[6];
    const float* ff2_b = (const float*)d_in[7];
    const float* ta_w  = (const float*)d_in[8];
    const float* ta_b  = (const float*)d_in[9];
    const float* tb_w  = (const float*)d_in[10];
    const float* tb_b  = (const float*)d_in[11];
    const float* out_w = (const float*)d_in[12];
    const float* out_b = (const float*)d_in[13];
    float* yout = (float*)d_out;

    dim3 grid(256 / MTILE);   // 16 batch tiles, one per WGP
    dim3 block(128);          // 4 waves of 32
    cfc_scan_kernel<<<grid, block, 0, stream>>>(
        x, h0, bb_w, bb_b, ff1_w, ff1_b, ff2_w, ff2_b,
        ta_w, ta_b, tb_w, tb_b, out_w, out_b, yout);
}
